// tet10_encoder_4346506903564
// MI455X (gfx1250) — matrix operand order repaired and checked by reading, not compiled, not run on hardware
//
#include <hip/hip_runtime.h>

typedef __attribute__((ext_vector_type(16))) _Float16 v16h;
typedef __attribute__((ext_vector_type(8)))  _Float16 v8h;
typedef __attribute__((ext_vector_type(8)))  float    v8f;

#define B_TOT 4096
#define T_LEN 512
#define I_IN  30
#define H_HID 16

static constexpr int WPB  = 4;   // waves per block (wave32)
static constexpr int LDSW = 24;  // _Float16 per LDS row: 48B rows, 16B-aligned b128 slices
static constexpr int PF   = 16;  // x prefetch distance in timesteps

__device__ __forceinline__ float fast_sigmoid(float x) {
    return __builtin_amdgcn_rcpf(1.0f + __expf(-x));
}
__device__ __forceinline__ float fast_tanh(float x) {
    float e = __expf(-2.0f * x);
    return (1.0f - e) * __builtin_amdgcn_rcpf(1.0f + e);
}
__device__ __forceinline__ v8f bcast8(float s) {
    v8f c;
#pragma unroll
    for (int r = 0; r < 8; ++r) c[r] = s;
    return c;
}

// Build B-matrix (K=32 x N=16, f16) from row-major W: B[k][n] = W[(goff+n)*ldw + k].
// CDNA5 B layout (wave32): lanes 0-15: N=lane, K in {0..7}U{16..23};
//                          lanes 16-31: N=lane-16, K in {8..15}U{24..31}. K >= kmax zero-padded.
__device__ __forceinline__ v16h load_B(const float* __restrict__ W, int goff, int ldw,
                                       int kmax, int lane) {
    const int half = lane >> 4, n = lane & 15;
    const float* row = W + (size_t)(goff + n) * ldw;
    v16h b;
#pragma unroll
    for (int v = 0; v < 8; ++v) {
        int k0 = 2 * v + half * 8 + (v >= 4 ? 8 : 0);
        int k1 = k0 + 1;
        int s0 = (k0 < kmax) ? k0 : 0;   // clamp address, select value: no divergent branch
        int s1 = (k1 < kmax) ? k1 : 0;
        float f0 = row[s0], f1 = row[s1];
        b[2 * v]     = (k0 < kmax) ? (_Float16)f0 : (_Float16)0.0f;
        b[2 * v + 1] = (k1 < kmax) ? (_Float16)f1 : (_Float16)0.0f;
    }
    return b;
}

// A-matrix tile of x for one timestep (M=16 x K=32 f16, K padded from I=30).
// A layout: lanes 0-15: M=lane, K in {0..7}U{16..23}; lanes 16-31: M=lane-16, K in {8..15}U{24..31}.
// Each lane reads only its own batch row -> xrow points at x[row, t, 0].
__device__ __forceinline__ v16h load_x_tile(const float* __restrict__ xrow, int half) {
    const int b0 = half * 8;   // 0 or 8   (always < 16, valid)
    const int b1 = b0 + 16;    // 16 or 24 (k=30,31 are pad)
    v16h a;
#pragma unroll
    for (int j = 0; j < 8; ++j) a[j] = (_Float16)xrow[b0 + j];
#pragma unroll
    for (int j = 0; j < 6; ++j) a[8 + j] = (_Float16)xrow[b1 + j];
#pragma unroll
    for (int j = 6; j < 8; ++j) {        // k = 22/23 (valid) or 30/31 (pad)
        int k  = b1 + j;
        int ks = (k < I_IN) ? k : 0;
        float v = xrow[ks];
        a[8 + j] = (k < I_IN) ? (_Float16)v : (_Float16)0.0f;
    }
    return a;
}

__global__ __launch_bounds__(WPB * 32)
void gru_head_fused(const float* __restrict__ x, const int* __restrict__ lengths,
                    const float* __restrict__ w_ih, const float* __restrict__ w_hh,
                    const float* __restrict__ b_ih, const float* __restrict__ b_hh,
                    const float* __restrict__ fc1_w, const float* __restrict__ fc1_b,
                    const float* __restrict__ fc2_w, const float* __restrict__ fc2_b,
                    const float* __restrict__ fc3_w, const float* __restrict__ fc3_b,
                    float* __restrict__ out) {
    __shared__ alignas(16) _Float16 hbuf[WPB][16][LDSW];  // per-wave h re-layout tile

    const int lane = threadIdx.x & 31;
    const int wid  = threadIdx.x >> 5;
    const int tile = blockIdx.x * WPB + wid;
    if (tile >= B_TOT / 16) return;          // wave-uniform
    const int half = lane >> 4;
    const int n    = lane & 15;
    const int rowM = tile * 16;              // batch base of this wave's tile

    _Float16* lds = &hbuf[wid][0][0];

    // ---- loop-invariant weights straight into B-matrix registers ----
    v16h Bih0 = load_B(w_ih,  0, I_IN,  I_IN,  lane);  // r gate
    v16h Bih1 = load_B(w_ih, 16, I_IN,  I_IN,  lane);  // z gate
    v16h Bih2 = load_B(w_ih, 32, I_IN,  I_IN,  lane);  // n gate
    v16h Bhh0 = load_B(w_hh,  0, H_HID, H_HID, lane);
    v16h Bhh1 = load_B(w_hh,  16, H_HID, H_HID, lane);
    v16h Bhh2 = load_B(w_hh,  32, H_HID, H_HID, lane);

    const float brz0 = b_ih[n]      + b_hh[n];        // r: b_ih + b_hh folded into x-proj C
    const float brz1 = b_ih[16 + n] + b_hh[16 + n];   // z
    const float bxn  = b_ih[32 + n];                  // n: x side
    const float bhn  = b_hh[32 + n];                  // n: h side (kept separate: r * (hWn + bhn))

    int lenm1[8];                                     // lengths-1 per D-layout row M = r + 8*half
#pragma unroll
    for (int r = 0; r < 8; ++r) lenm1[r] = lengths[rowM + r + half * 8] - 1;

    v8f h = bcast8(0.0f), last = bcast8(0.0f);
#pragma unroll
    for (int r = 0; r < 8; ++r) lds[(r + half * 8) * LDSW + n] = (_Float16)0.0f;  // h0 = 0

    const float* xbase = x + (size_t)(rowM + n) * T_LEN * I_IN;  // lane's batch row
    v16h ax = load_x_tile(xbase, half);

    for (int t = 0; t < T_LEN; ++t) {
        if (t + PF < T_LEN) __builtin_prefetch(xbase + (size_t)(t + PF) * I_IN, 0, 1);
        v16h axn = ax;
        if (t + 1 < T_LEN) axn = load_x_tile(xbase + (size_t)(t + 1) * I_IN, half);

        // x-side projections, bias pre-loaded into C (K = 30 -> one 16x16x32 WMMA per gate)
        v8f cr  = __builtin_amdgcn_wmma_f32_16x16x32_f16(false, ax, false, Bih0, (short)0, bcast8(brz0), false, false);
        v8f cz  = __builtin_amdgcn_wmma_f32_16x16x32_f16(false, ax, false, Bih1, (short)0, bcast8(brz1), false, false);
        v8f cnx = __builtin_amdgcn_wmma_f32_16x16x32_f16(false, ax, false, Bih2, (short)0, bcast8(bxn),  false, false);

        // h (D layout, prev step) -> A layout via per-wave LDS tile: one b128 per lane
        v8h hlo = *(const v8h*)(lds + n * LDSW + half * 8);
        v16h ah;
#pragma unroll
        for (int i = 0; i < 8; ++i) { ah[i] = hlo[i]; ah[8 + i] = (_Float16)0.0f; }

        cr = __builtin_amdgcn_wmma_f32_16x16x32_f16(false, ah, false, Bhh0, (short)0, cr, false, false);
        cz = __builtin_amdgcn_wmma_f32_16x16x32_f16(false, ah, false, Bhh1, (short)0, cz, false, false);
        v8f cnh = __builtin_amdgcn_wmma_f32_16x16x32_f16(false, ah, false, Bhh2, (short)0, bcast8(bhn), false, false);

        // GRU pointwise update directly in D layout; capture h at t == len-1 (cndmask)
#pragma unroll
        for (int r = 0; r < 8; ++r) {
            float rg = fast_sigmoid(cr[r]);
            float zg = fast_sigmoid(cz[r]);
            float ng = fast_tanh(cnx[r] + rg * cnh[r]);
            float hn = (1.0f - zg) * ng + zg * h[r];
            h[r] = hn;
            last[r] = (t == lenm1[r]) ? hn : last[r];
        }
#pragma unroll
        for (int r = 0; r < 8; ++r) lds[(r + half * 8) * LDSW + n] = (_Float16)h[r];

        ax = axn;
    }

    // ---- 3x FC16x16 head: relu between layers, same LDS re-layout trick ----
    const float* fw[3] = {fc1_w, fc2_w, fc3_w};
    const float* fb[3] = {fc1_b, fc2_b, fc3_b};
    v8f cur = last;
#pragma unroll
    for (int L = 0; L < 3; ++L) {
#pragma unroll
        for (int r = 0; r < 8; ++r) {
            float v = cur[r];
            if (L > 0) v = fmaxf(v, 0.0f);   // relu on inputs of fc2/fc3 only
            lds[(r + half * 8) * LDSW + n] = (_Float16)v;
        }
        v8h lo = *(const v8h*)(lds + n * LDSW + half * 8);
        v16h ae;
#pragma unroll
        for (int i = 0; i < 8; ++i) { ae[i] = lo[i]; ae[8 + i] = (_Float16)0.0f; }
        v16h Bf = load_B(fw[L], 0, H_HID, H_HID, lane);
        cur = __builtin_amdgcn_wmma_f32_16x16x32_f16(false, ae, false, Bf, (short)0, bcast8(fb[L][n]), false, false);
    }

    // out[b][n], b = rowM + r + 8*half; lanes 0..15 cover consecutive n -> coalesced
#pragma unroll
    for (int r = 0; r < 8; ++r)
        out[(size_t)(rowM + r + half * 8) * H_HID + n] = cur[r];
}

extern "C" void kernel_launch(void* const* d_in, const int* in_sizes, int n_in,
                              void* d_out, int out_size, void* d_ws, size_t ws_size,
                              hipStream_t stream) {
    (void)in_sizes; (void)n_in; (void)d_ws; (void)ws_size; (void)out_size;
    const float* x       = (const float*)d_in[0];
    const int*   lengths = (const int*)  d_in[1];
    const float* w_ih    = (const float*)d_in[2];
    const float* w_hh    = (const float*)d_in[3];
    const float* b_ih    = (const float*)d_in[4];
    const float* b_hh    = (const float*)d_in[5];
    const float* fc1_w   = (const float*)d_in[6];
    const float* fc1_b   = (const float*)d_in[7];
    const float* fc2_w   = (const float*)d_in[8];
    const float* fc2_b   = (const float*)d_in[9];
    const float* fc3_w   = (const float*)d_in[10];
    const float* fc3_b   = (const float*)d_in[11];

    dim3 grid((B_TOT / 16) / WPB);   // 64 blocks x 4 waves = 256 waves = B/16 tiles
    dim3 block(WPB * 32);
    gru_head_fused<<<grid, block, 0, stream>>>(x, lengths, w_ih, w_hh, b_ih, b_hh,
                                               fc1_w, fc1_b, fc2_w, fc2_b, fc3_w, fc3_b,
                                               (float*)d_out);
}
